// BiGRUEncoder_73701638799693
// MI455X (gfx1250) — compile-verified, matrix operand order checked by reading, not run on hardware
//
#include <hip/hip_runtime.h>

// ---------------- problem constants ----------------
constexpr int B   = 32;
constexpr int T   = 512;
constexpr int NIN = 2049;
constexpr int F   = 1024;
constexpr int L   = 10;
constexpr int OUT_T = T - 2 * L;       // 492
constexpr int G3F = 3 * F;             // 3072

// ---------------- WMMA types (CDNA5 / gfx1250, wave32) ----------------
typedef __bf16 bf16x16 __attribute__((ext_vector_type(16)));
typedef float  f32x8   __attribute__((ext_vector_type(8)));

union Frag16 { bf16x16 v; int4 q[2]; };   // 32 bytes: two 16B chunks per lane

static __device__ inline f32x8 wmma_bf16(bf16x16 a, bf16x16 b, f32x8 c) {
  // D = A(16x32 bf16) x B(32x16 bf16) + C(16x16 f32)
  return __builtin_amdgcn_wmma_f32_16x16x32_bf16(
      /*neg_a=*/false, a, /*neg_b=*/false, b,
      /*c_mod=*/(short)0, c, /*reuse_a=*/false, /*reuse_b=*/false);
}

static __device__ inline unsigned short f32_to_bf16(float x) {
  unsigned u = __float_as_uint(x);
  unsigned r = u + 0x7FFFu + ((u >> 16) & 1u);   // round to nearest even
  return (unsigned short)(r >> 16);
}

// WGP-scope prefetch: fills all cache levels incl. WGP$ (scope 0), unlike
// __builtin_prefetch which lowers to SE/DEV scope (GL2-only) on gfx1250.
static __device__ inline void prefetch_wgp(const void* p) {
  asm volatile("global_prefetch_b8 %0, off" :: "v"(p));
}

// ---------------- prep kernels ----------------
__global__ void k_zero_bar(unsigned* __restrict__ bar) {
  if (threadIdx.x < 64) bar[threadIdx.x] = 0u;
}

__global__ void k_f32_to_bf16(const float* __restrict__ src,
                              unsigned short* __restrict__ dst, int n) {
  for (int i = blockIdx.x * blockDim.x + threadIdx.x; i < n;
       i += gridDim.x * blockDim.x)
    dst[i] = f32_to_bf16(src[i]);
}

// input_x [B,T,2049] -> xbf [B,T,1024] bf16 (ALPHA == 1.0 so x**1 == x)
__global__ void k_extract_x(const float* __restrict__ x,
                            unsigned short* __restrict__ xbf) {
  const long n = (long)B * T * F;
  for (long i = blockIdx.x * (long)blockDim.x + threadIdx.x; i < n;
       i += (long)gridDim.x * blockDim.x) {
    long bt = i >> 10;                 // /F
    int  f  = (int)(i & (F - 1));
    xbf[i] = f32_to_bf16(x[bt * NIN + f]);
  }
}

// ---------------- persistent bidirectional GRU scan ----------------
// Grid: 32 WGs = 2 dirs x 16 column tiles (64 F-cols each); 256 threads = 8 waves.
// Wave w: m = w>>2 (batch tile of 16), fsub = w&3 (16-col sub-tile).
// Per step: stage A=[x_t | h_{t-1}] (32 x 2048 bf16) in LDS, then two
// software-pipelined 32-iter WMMA phases (x-part over Wih, h-part over Whh).
// Weight loads use UNIFORM base + divergent 32-bit int4 index so the backend
// emits global_load_b128 v, vIdx, s[base] scale_offset offset:imm.
__global__ __launch_bounds__(256)
void bigru_recurrence(const unsigned short* __restrict__ xbf,
                      const unsigned short* __restrict__ wih,   // [2][3F][F] bf16
                      const unsigned short* __restrict__ whh,   // [2][3F][F] bf16
                      const float* __restrict__ bih_f, const float* __restrict__ bhh_f,
                      const float* __restrict__ bih_b, const float* __restrict__ bhh_b,
                      const float* __restrict__ input_x,
                      float* __restrict__ out,
                      unsigned short* __restrict__ hbf,   // [2][2][32][F] bf16 ping-pong
                      float* __restrict__ hf32,           // [2][2][32][F] f32 ping-pong
                      unsigned* __restrict__ bar) {
  extern __shared__ int4 Aq[];                // [32 rows][AQS int4], padded rows
  constexpr int AQS = 257;                    // 2048 els + 8 pad -> 4112B = 257*16B

  const int tid  = threadIdx.x;
  const int lane = tid & 31;
  const int wave = tid >> 5;
  const int m    = wave >> 2;                 // 0..1 (batch tile)
  const int fsub = wave & 3;                  // 0..3 (16-col sub-tile)
  const int lr   = lane & 15;
  const int half = lane >> 4;
  const int dir  = blockIdx.x >> 4;           // 0=fwd, 1=bwd
  const int ctil = blockIdx.x & 15;
  const int fcol = ctil * 64 + fsub * 16 + lr;

  // UNIFORM weight bases (per direction), int4 units; rows are 128 int4 wide.
  const int4* wihq = (const int4*)wih + (size_t)dir * (G3F * (F / 8));
  const int4* whhq = (const int4*)whh + (size_t)dir * (G3F * (F / 8));
  const int4* xq   = (const int4*)xbf;
  const int4* hq   = (const int4*)hbf;
  const float* bih = dir ? bih_b : bih_f;
  const float* bhh = dir ? bhh_b : bhh_f;

  unsigned* cnt = bar + dir * 32;             // 64B-separated per direction
  unsigned* gen = bar + dir * 32 + 16;

  const float bir = bih[fcol],         bhr = bhh[fcol];
  const float biz = bih[F + fcol],     bhz = bhh[F + fcol];
  const float bin = bih[2 * F + fcol], bhn = bhh[2 * F + fcol];

  const int aRow = (m * 16 + lr) * AQS;

  for (int s = 0; s < T; ++s) {
    const int p = s & 1;
    const int t = dir ? (T - 1 - s) : s;

    // Launder a zero-valued k-offset per step through an SGPR: indices become
    // step-variant (no cross-step hoist/spill of the weight loads), stay
    // uniform in their zk part, and bases remain uniform SGPR pairs.
    int zk = 0;
    asm volatile("" : "+s"(zk));
    // divergent 32-bit int4 indices for the three gate rows (shared by both
    // weight arrays: identical [3F][F] layout)
    const int iR = (fcol)         * 128 + half + zk;
    const int iZ = (F + fcol)     * 128 + half + zk;
    const int iN = (2 * F + fcol) * 128 + half + zk;

    // Warm WGP$ for the first k-groups of each stream (fire-and-forget,
    // overlaps the LDS staging below).
    prefetch_wgp(wihq + iR); prefetch_wgp(wihq + iZ); prefetch_wgp(wihq + iN);
    prefetch_wgp(whhq + iR); prefetch_wgp(whhq + iZ); prefetch_wgp(whhq + iN);

    // ---- stage A = [x_t | h_{s-1}] into LDS (bf16) ----
#pragma unroll 4
    for (int i = tid; i < 32 * 128; i += 256) {
      int row = i >> 7, c = i & 127;
      Aq[row * AQS + c] = xq[((size_t)row * T + t) * 128 + c];
    }
    if (s == 0) {
      const int4 z4 = make_int4(0, 0, 0, 0);
#pragma unroll 4
      for (int i = tid; i < 32 * 128; i += 256) {
        int row = i >> 7, c = i & 127;
        Aq[row * AQS + 128 + c] = z4;
      }
    } else {
#pragma unroll 4
      for (int i = tid; i < 32 * 128; i += 256) {
        int row = i >> 7, c = i & 127;
        Aq[row * AQS + 128 + c] = hq[((size_t)(dir * 2 + p) * 32 + row) * 128 + c];
      }
    }

    // ---- pre-issue combine-stage operands (independent of the GEMM) ----
    float hold[8], xres[8];
#pragma unroll
    for (int i = 0; i < 8; ++i) {
      const int row = m * 16 + half * 8 + i;
      xres[i] = input_x[((size_t)row * T + t) * NIN + fcol];
      hold[i] = (s == 0) ? 0.f
              : hf32[((size_t)(dir * 2 + p) * 32 + row) * F + fcol];
    }
    __syncthreads();

    f32x8 c_r = {}, c_z = {}, c_in = {}, c_hn = {};

    // One phase = 32 k-iters over one weight array (uniform base Wq).
    // B streams software-pipelined one iter ahead; per k-group only four
    // 32-bit index adds, all loads use immediate offsets.
    auto gemm_phase = [&](const int4* Wq, int aBase, f32x8& cn)
        __attribute__((always_inline)) {
      int aq  = aRow + aBase + half;
      int iRk = iR, iZk = iZ, iNk = iN;
      Frag16 br, bz, bn;
      br.q[0] = Wq[iRk];  br.q[1] = Wq[iRk + 2];
      bz.q[0] = Wq[iZk];  bz.q[1] = Wq[iZk + 2];
      bn.q[0] = Wq[iNk];  bn.q[1] = Wq[iNk + 2];
      for (int ko = 0; ko < 8; ++ko) {
#pragma unroll
        for (int ki = 0; ki < 4; ++ki) {
          const int oa = ki * 4;         // current A offset within group
          const int ob = ki * 4 + 4;     // look-ahead B offset within group
          Frag16 a, br2, bz2, bn2;
          a.q[0]   = Aq[aq + oa];       a.q[1]   = Aq[aq + oa + 2];
          br2.q[0] = Wq[iRk + ob];      br2.q[1] = Wq[iRk + ob + 2];
          bz2.q[0] = Wq[iZk + ob];      bz2.q[1] = Wq[iZk + ob + 2];
          bn2.q[0] = Wq[iNk + ob];      bn2.q[1] = Wq[iNk + ob + 2];
          c_r = wmma_bf16(a.v, br.v, c_r);
          c_z = wmma_bf16(a.v, bz.v, c_z);
          cn  = wmma_bf16(a.v, bn.v, cn);
          br = br2; bz = bz2; bn = bn2;
        }
        aq += 16; iRk += 16; iZk += 16; iNk += 16;
      }
    };
    gemm_phase(wihq, 0,   c_in);   // x-part: K = 0..1023
    gemm_phase(whhq, 128, c_hn);   // h-part: K = 1024..2047

    // ---- gates + residual + state update (C elem i -> M = m*16+8*half+i, N = lr) ----
#pragma unroll
    for (int i = 0; i < 8; ++i) {
      const int row = m * 16 + half * 8 + i;            // batch row 0..31
      float rg = 1.f / (1.f + __expf(-(c_r[i] + bir + bhr)));
      float zg = 1.f / (1.f + __expf(-(c_z[i] + biz + bhz)));
      float ng = tanhf(c_in[i] + bin + rg * (c_hn[i] + bhn));
      float hnew = (1.f - zg) * ng + zg * hold[i] + xres[i];
      const size_t hoff = ((size_t)(dir * 2 + (p ^ 1)) * 32 + row) * F + fcol;
      hf32[hoff] = hnew;
      hbf[hoff]  = f32_to_bf16(hnew);
      if (s >= L && s < T - L)
        out[(size_t)row * (OUT_T * 2 * F) + (size_t)(s - L) * (2 * F) +
            dir * F + fcol] = hnew;
    }

    // ---- device-scope generation barrier across the 16 WGs of this direction ----
    __threadfence();
    __syncthreads();
    if (tid == 0) {
      unsigned g = __hip_atomic_load(gen, __ATOMIC_RELAXED, __HIP_MEMORY_SCOPE_AGENT);
      unsigned a = __hip_atomic_fetch_add(cnt, 1u, __ATOMIC_ACQ_REL, __HIP_MEMORY_SCOPE_AGENT);
      if (a == 15u) {
        __hip_atomic_store(cnt, 0u, __ATOMIC_RELEASE, __HIP_MEMORY_SCOPE_AGENT);
        __hip_atomic_fetch_add(gen, 1u, __ATOMIC_RELEASE, __HIP_MEMORY_SCOPE_AGENT);
      } else {
        while (__hip_atomic_load(gen, __ATOMIC_ACQUIRE, __HIP_MEMORY_SCOPE_AGENT) == g)
          __builtin_amdgcn_s_sleep(2);
      }
    }
    __syncthreads();
  }
}

// ---------------- host-side launch ----------------
extern "C" void kernel_launch(void* const* d_in, const int* in_sizes, int n_in,
                              void* d_out, int out_size, void* d_ws, size_t ws_size,
                              hipStream_t stream) {
  (void)in_sizes; (void)n_in; (void)out_size; (void)ws_size;
  const float* input_x = (const float*)d_in[0];
  const float* Wih_f   = (const float*)d_in[1];
  const float* Whh_f   = (const float*)d_in[2];
  const float* bih_f   = (const float*)d_in[3];
  const float* bhh_f   = (const float*)d_in[4];
  const float* Wih_b   = (const float*)d_in[5];
  const float* Whh_b   = (const float*)d_in[6];
  const float* bih_b   = (const float*)d_in[7];
  const float* bhh_b   = (const float*)d_in[8];
  float* out = (float*)d_out;

  // carve workspace (~60 MB)
  char* ws = (char*)d_ws;
  size_t off = 0;
  auto carve = [&](size_t bytes) -> char* {
    char* p = ws + off;
    off += (bytes + 255) & ~(size_t)255;
    return p;
  };
  unsigned short* xbf = (unsigned short*)carve((size_t)B * T * F * 2);      // 33.5 MB
  unsigned short* wih = (unsigned short*)carve((size_t)2 * G3F * F * 2);    // 12.6 MB
  unsigned short* whh = (unsigned short*)carve((size_t)2 * G3F * F * 2);    // 12.6 MB
  unsigned short* hbf = (unsigned short*)carve((size_t)2 * 2 * B * F * 2);  // 256 KB
  float*          hfp = (float*)carve((size_t)2 * 2 * B * F * 4);           // 512 KB
  unsigned*       bar = (unsigned*)carve(256);

  k_zero_bar<<<1, 64, 0, stream>>>(bar);

  const int nW = G3F * F;
  k_f32_to_bf16<<<2048, 256, 0, stream>>>(Wih_f, wih,      nW);
  k_f32_to_bf16<<<2048, 256, 0, stream>>>(Wih_b, wih + nW, nW);
  k_f32_to_bf16<<<2048, 256, 0, stream>>>(Whh_f, whh,      nW);
  k_f32_to_bf16<<<2048, 256, 0, stream>>>(Whh_b, whh + nW, nW);
  k_extract_x<<<4096, 256, 0, stream>>>(input_x, xbf);

  const size_t smem = (size_t)32 * 257 * sizeof(int4);   // ~128.5 KB dynamic LDS
  bigru_recurrence<<<dim3(32), dim3(256), smem, stream>>>(
      xbf, wih, whh, bih_f, bhh_f, bih_b, bhh_b,
      input_x, out, hbf, hfp, bar);
}
